// EGNN_Net_17815524344059
// MI455X (gfx1250) — matrix operand order repaired and checked by reading, not compiled, not run on hardware
//
#include <hip/hip_runtime.h>
#include <hip/hip_bf16.h>
#include <math.h>

typedef __attribute__((ext_vector_type(16))) __bf16 v16bf;
typedef __attribute__((ext_vector_type(8)))  float  v8f;
typedef unsigned short u16;
typedef unsigned int   u32;
typedef unsigned long long u64;

#define N_REC 500
#define N_LIG 300
#define NTOT  800
#define LM_DIM 1280
#define NODE_DIM 128
#define EDGE_DIM 64
#define POS_DIM 66
#define KNN 20
#define SAMPLE 40
#define DEG 60
#define NEDGE (NTOT*DEG)
#define K1 321
#define K1P 352
#define KE 257
#define KEP 288
#define NPAIR (N_REC*N_LIG)
#define TAILW 96   /* per-edge packed tail: radial(1) + ea(64) + pad(31) bf16 = 192B */

// ---------------- small math helpers ----------------
struct F3 { float x, y, z; };
__device__ __forceinline__ F3 mk3(const float* p){ return {p[0],p[1],p[2]}; }
__device__ __forceinline__ F3 sub3(F3 a, F3 b){ return {a.x-b.x, a.y-b.y, a.z-b.z}; }
__device__ __forceinline__ F3 add3(F3 a, F3 b){ return {a.x+b.x, a.y+b.y, a.z+b.z}; }
__device__ __forceinline__ F3 scl3(F3 a, float s){ return {a.x*s, a.y*s, a.z*s}; }
__device__ __forceinline__ float dot3(F3 a, F3 b){ return a.x*b.x + a.y*b.y + a.z*b.z; }
__device__ __forceinline__ F3 cross3(F3 a, F3 b){
  return {a.y*b.z-a.z*b.y, a.z*b.x-a.x*b.z, a.x*b.y-a.y*b.x};
}
__device__ __forceinline__ float norm3(F3 a){ return sqrtf(dot3(a,a)); }

__device__ __forceinline__ float siluf(float x){ return x / (1.0f + expf(-x)); }

__device__ __forceinline__ u16 f2bf(float f){
  u32 u = __float_as_uint(f);
  u32 r = u + 0x7fffu + ((u >> 16) & 1u);   // round-to-nearest-even
  return (u16)(r >> 16);
}

__device__ __forceinline__ float dihedral_deg(F3 p0, F3 p1, F3 p2, F3 p3){
  F3 b0 = sub3(p0,p1), b1 = sub3(p2,p1), b2 = sub3(p3,p2);
  float inv = 1.0f / (norm3(b1) + 1e-8f);
  b1 = scl3(b1, inv);
  F3 v = sub3(b0, scl3(b1, dot3(b0,b1)));
  F3 w = sub3(b2, scl3(b1, dot3(b2,b1)));
  float xx = dot3(v,w);
  float yy = dot3(cross3(b1,v), w);
  return atan2f(yy,xx) * 57.29577951308232f;
}
__device__ __forceinline__ float planar_deg(F3 a, F3 b, F3 c){
  F3 v = sub3(a,b), w = sub3(c,b);
  float cs = dot3(v,w) / (norm3(v)*norm3(w) + 1e-8f);
  cs = fminf(1.0f, fmaxf(-1.0f, cs));
  return acosf(cs) * 57.29577951308232f;
}
__device__ __forceinline__ int bin_of(float x, float lo, float hi, int nb){
  int c = 0; float step = (hi - lo) / (float)(nb - 2);
  for (int i = 0; i < nb-1; ++i) c += (x > (lo + i*step)) ? 1 : 0;
  return c;
}
__device__ __forceinline__ float gumbel_hash(u32 x){
  x ^= 0x9E3779B9u; x *= 0x85EBCA6Bu; x ^= x >> 13; x *= 0xC2B2AE35u; x ^= x >> 16;
  float u = ((x >> 8) + 0.5f) * (1.0f/16777216.0f);   // (0,1)
  return -logf(-logf(u));
}

// ---------------- CDNA5 async global->LDS copy helpers ----------------
// VDST VGPR holds the wave LDS byte offset (= low 32 bits of a generic pointer
// to __shared__); VADDR is the 64-bit global byte address.  Tracked by ASYNCcnt.
__device__ __forceinline__ u32 lds_off(const void* p){
  return (u32)(u64)p;
}
__device__ __forceinline__ void async_copy16(u32 lds_byte_off, const void* gaddr){
  asm volatile("global_load_async_to_lds_b128 %0, %1, off"
               :: "v"(lds_byte_off), "v"((u64)gaddr) : "memory");
}
__device__ __forceinline__ void async_wait0(){
  asm volatile("s_wait_asynccnt 0" ::: "memory");
}

// ---------------- WMMA fragment helpers (bf16, wave32) ----------------
union FragBF { v16bf v; u32 u[8]; };

// A: 16x32 tile, row-major [m][k], lda elements (even). ISA 16-bit A layout.
__device__ __forceinline__ v16bf load_a_frag(const u16* A, int lda, int lane){
  int m = lane & 15, half = lane >> 4;
  const u32* row = (const u32*)(A + m*lda);
  FragBF f;
#pragma unroll
  for (int j = 0; j < 8; ++j){
    int k = 2*(j&3) + ((j&4) ? 16 : 0) + 8*half;
    f.u[j] = row[k >> 1];
  }
  return f.v;
}
// B: 32x16 tile stored as Bs[n][k] ([N][K] row-major), ldb elements (even).
__device__ __forceinline__ v16bf load_b_frag(const u16* B, int ldb, int lane){
  int n = lane & 15, half = lane >> 4;
  const u32* col = (const u32*)(B + n*ldb);
  FragBF f;
#pragma unroll
  for (int j = 0; j < 8; ++j){
    int k = 2*(j&3) + ((j&4) ? 16 : 0) + 8*half;
    f.u[j] = col[k >> 1];
  }
  return f.v;
}
__device__ __forceinline__ v8f wmma_bf16(v16bf a, v16bf b, v8f c){
  return __builtin_amdgcn_wmma_f32_16x16x32_bf16(false, a, false, b, (short)0, c, false, false);
}
__device__ __forceinline__ v8f v8zero(){ v8f z = {0.f,0.f,0.f,0.f,0.f,0.f,0.f,0.f}; return z; }

// ---------------- prep / conversion kernels ----------------
__global__ void init_scalars_kernel(float* scal){
  if (blockIdx.x == 0 && threadIdx.x == 0){ scal[0] = 0.0f; scal[1] = 0.0f; }
}

__global__ void convert_pad_kernel(const float* __restrict__ src, u16* __restrict__ dst,
                                   int R, int C, int Cpad){
  int idx = blockIdx.x*256 + threadIdx.x;
  if (idx >= R*Cpad) return;
  int r = idx / Cpad, c = idx - r*Cpad;
  dst[idx] = (c < C) ? f2bf(src[(size_t)r*C + c]) : (u16)0;
}

__global__ void prep_geom_kernel(const float* __restrict__ rec_pos, const float* __restrict__ lig_pos,
                                 float* __restrict__ nt, float* __restrict__ ca, float* __restrict__ cb){
  int i = blockIdx.x*256 + threadIdx.x;
  if (i >= NTOT) return;
  const float* p = (i < N_REC) ? (rec_pos + (size_t)i*9) : (lig_pos + (size_t)(i-N_REC)*9);
  F3 Nt = {p[0],p[1],p[2]}, Ca = {p[3],p[4],p[5]}, Ct = {p[6],p[7],p[8]};
  F3 b = sub3(Ca,Nt), c = sub3(Ct,Ca);
  F3 a = cross3(b,c);
  F3 Cb = add3(add3(scl3(a,-0.58273431f), scl3(b,0.56802827f)),
               add3(scl3(c,-0.54067466f), Ca));
  nt[i*3+0]=Nt.x; nt[i*3+1]=Nt.y; nt[i*3+2]=Nt.z;
  ca[i*3+0]=Ca.x; ca[i*3+1]=Ca.y; ca[i*3+2]=Ca.z;
  cb[i*3+0]=Cb.x; cb[i*3+1]=Cb.y; cb[i*3+2]=Cb.z;
}

__global__ void dm_kernel(const float* __restrict__ ca, float* __restrict__ dm){
  int idx = blockIdx.x*256 + threadIdx.x;
  if (idx >= NTOT*NTOT) return;
  int i = idx / NTOT, j = idx - i*NTOT;
  float dx = ca[i*3+0]-ca[j*3+0], dy = ca[i*3+1]-ca[j*3+1], dz = ca[i*3+2]-ca[j*3+2];
  dm[idx] = sqrtf(fmaxf(dx*dx + dy*dy + dz*dz, 0.0f));
}

// ---------------- graph construction: KNN + gumbel sampling ----------------
__global__ void __launch_bounds__(256) knn_sample_kernel(const float* __restrict__ dm,
                                                         int* __restrict__ dsti){
  __shared__ float d[NTOT];
  __shared__ float key[NTOT];
  __shared__ int   sel[NTOT];
  __shared__ float rv[256];
  __shared__ int   ri[256];
  int i = blockIdx.x, t = threadIdx.x;
  for (int j = t; j < NTOT; j += 256){ d[j] = dm[(size_t)i*NTOT + j]; sel[j] = 0; }
  __syncthreads();
  for (int pick = 0; pick < KNN; ++pick){
    float bv = 3.4e38f; int bi = NTOT;
    for (int j = t; j < NTOT; j += 256){
      if (!sel[j]){ float v = d[j]; if (v < bv || (v == bv && j < bi)){ bv = v; bi = j; } }
    }
    rv[t] = bv; ri[t] = bi; __syncthreads();
    for (int s = 128; s > 0; s >>= 1){
      if (t < s){
        float ov = rv[t+s]; int oi = ri[t+s];
        if (ov < rv[t] || (ov == rv[t] && oi < ri[t])){ rv[t] = ov; ri[t] = oi; }
      }
      __syncthreads();
    }
    if (t == 0){ dsti[i*DEG + pick] = ri[0]; sel[ri[0]] = 1; }
    __syncthreads();
  }
  for (int j = t; j < NTOT; j += 256){
    float k;
    if (sel[j]) k = -3.4e38f;
    else {
      float dd = fmaxf(d[j], 1e-10f);
      k = -3.0f*logf(dd) + gumbel_hash((u32)(i*NTOT + j)*2654435761u + 42u);
    }
    key[j] = k;
  }
  __syncthreads();
  for (int pick = 0; pick < SAMPLE; ++pick){
    float bv = -3.4e38f; int bi = NTOT;
    for (int j = t; j < NTOT; j += 256){
      float v = key[j]; if (v > bv || (v == bv && j < bi)){ bv = v; bi = j; }
    }
    rv[t] = bv; ri[t] = bi; __syncthreads();
    for (int s = 128; s > 0; s >>= 1){
      if (t < s){
        float ov = rv[t+s]; int oi = ri[t+s];
        if (ov > rv[t] || (ov == rv[t] && oi < ri[t])){ rv[t] = ov; ri[t] = oi; }
      }
      __syncthreads();
    }
    if (t == 0){ dsti[i*DEG + KNN + pick] = ri[0]; key[ri[0]] = -3.4e38f; }
    __syncthreads();
  }
}

// ---------------- per-edge features: spatial one-hot gather + positional embed ----------------
__global__ void __launch_bounds__(64) edge_attr_kernel(const int* __restrict__ dsti,
    const float* __restrict__ nt, const float* __restrict__ ca, const float* __restrict__ cb,
    const float* __restrict__ pm, const float* __restrict__ spW, const float* __restrict__ posW,
    float* __restrict__ ea, float* __restrict__ radial){
  __shared__ int sb[4];
  __shared__ float srad;
  int e = blockIdx.x, t = threadIdx.x;
  int i = e / DEG;
  int j = dsti[e];
  if (t == 0){
    F3 Ci = mk3(ca + i*3), Cj = mk3(ca + j*3);
    F3 Bi = mk3(cb + i*3), Bj = mk3(cb + j*3);
    F3 Ni = mk3(nt + i*3);
    F3 dca = sub3(Ci, Cj);
    srad = dot3(dca, dca);
    float dist = norm3(sub3(Bi, Bj));
    bool msk = (dist < 20.0f) && (i != j);
    int db = bin_of(dist, 3.25f, 50.75f, 40);
    int ob = 0, tb = 0, pb = 0;
    if (msk){
      ob = bin_of(dihedral_deg(Ci, Bi, Bj, Cj), -180.0f, 180.0f, 24);
      tb = bin_of(dihedral_deg(Ni, Ci, Bi, Bj), -180.0f, 180.0f, 24);
      pb = bin_of(planar_deg(Ci, Bi, Bj), 0.0f, 180.0f, 12);
    }
    sb[0] = db; sb[1] = 40 + ob; sb[2] = 64 + tb; sb[3] = 88 + pb;
  }
  __syncthreads();
  float acc = spW[t*100 + sb[0]] + spW[t*100 + sb[1]] + spW[t*100 + sb[2]] + spW[t*100 + sb[3]];
  const float* pmrow = pm + ((size_t)i*NTOT + j)*POS_DIM;
  for (int k = 0; k < POS_DIM; ++k) acc += pmrow[k]*posW[t*POS_DIM + k];
  ea[(size_t)e*EDGE_DIM + t] = acc;
  if (t == 0) radial[e] = srad;
}

// pack layer-invariant edge tail: [radial | ea(64) | zeros] as 96 bf16 (192B, 16B aligned)
__global__ void __launch_bounds__(96) pack_tail_kernel(const float* __restrict__ radial,
    const float* __restrict__ ea, u16* __restrict__ tail){
  int e = blockIdx.x, t = threadIdx.x;
  u16 v = 0;
  if (t == 0)       v = f2bf(radial[e]);
  else if (t <= EDGE_DIM) v = f2bf(ea[(size_t)e*EDGE_DIM + (t-1)]);
  tail[(size_t)e*TAILW + t] = v;
}

// ---------------- node embedding GEMM: h = x @ Wse^T ----------------
__global__ void __launch_bounds__(256) embed_kernel(const u16* __restrict__ xb,
    const u16* __restrict__ WB, float* __restrict__ h, u16* __restrict__ hb){
  __shared__ __align__(16) u16 At[64*32];
  int tid = threadIdx.x, wid = tid >> 5, lane = tid & 31;
  int row0 = blockIdx.x*64;
  int r = tid >> 2, quad = tid & 3;
  int grow = row0 + r;
  bool valid = grow < NTOT;
  u32 lchunk = lds_off(At) + tid*16;          // each thread owns one 16B chunk / step
  if (!valid){                                 // rows past 800 stay zero for all kt
    u32* z = (u32*)At + tid*4;
    z[0]=0; z[1]=0; z[2]=0; z[3]=0;
  }
  __builtin_prefetch(WB + (size_t)(wid*16)*LM_DIM, 0, 0);
  v8f acc[4];
#pragma unroll
  for (int mt = 0; mt < 4; ++mt) acc[mt] = v8zero();
  for (int kt = 0; kt < LM_DIM/32; ++kt){
    if (valid) async_copy16(lchunk, xb + (size_t)grow*LM_DIM + kt*32 + quad*8);
    async_wait0();
    __syncthreads();
    v16bf b = load_b_frag(WB + (size_t)(wid*16)*LM_DIM + kt*32, LM_DIM, lane);
#pragma unroll
    for (int mt = 0; mt < 4; ++mt){
      v16bf a = load_a_frag(At + mt*16*32, 32, lane);
      acc[mt] = wmma_bf16(a, b, acc[mt]);
    }
    __syncthreads();
  }
  int half = lane >> 4, n = wid*16 + (lane & 15);
#pragma unroll
  for (int mt = 0; mt < 4; ++mt){
#pragma unroll
    for (int r8 = 0; r8 < 8; ++r8){
      int row = row0 + mt*16 + r8 + 8*half;
      if (row < NTOT){
        float v = acc[mt][r8];
        h[(size_t)row*NODE_DIM + n]  = v;
        hb[(size_t)row*NODE_DIM + n] = f2bf(v);
      }
    }
  }
}

// ---------------- fused EGCL edge pipeline ----------------
__global__ void __launch_bounds__(256) egcl_edge_kernel(const u16* __restrict__ hb,
    const int* __restrict__ dsti, const u16* __restrict__ tail,
    const u16* __restrict__ Wb1, const float* __restrict__ e1b,
    const u16* __restrict__ Wb2, const float* __restrict__ e2b,
    const float* __restrict__ attW, const float* __restrict__ attb,
    float* __restrict__ agg){
  __shared__ __align__(16) char smem[64*K1P*2 + 64*NODE_DIM*2];  // 45056 + 16384
  __shared__ float s_scale[64];
  __shared__ int   sdst[DEG];
  u16*   A1 = (u16*)smem;                         // 64 x 352 bf16
  u16*   M1 = (u16*)(smem + 64*K1P*2);            // 64 x 128 bf16
  float* M2 = (float*)smem;                       // 64 x 128 f32, overlays A1
  int i = blockIdx.x, tid = threadIdx.x, wid = tid >> 5, lane = tid & 31;

  __builtin_prefetch(Wb1 + (size_t)(wid*16)*K1P, 0, 0);
  __builtin_prefetch(Wb2 + (size_t)(wid*16)*NODE_DIM, 0, 0);

  if (tid < DEG) sdst[tid] = dsti[i*DEG + tid];
  // zero pad rows 60..63 (4 * 704B)
  for (int idx = tid; idx < 4*K1P/2; idx += 256) ((u32*)(A1 + DEG*K1P))[idx] = 0;
  __syncthreads();

  // async-stage A1 = [h_src | h_dst | tail], 60 rows x 44 x 16B chunks
  const int CH = 44;
  for (int q = tid; q < DEG*CH; q += 256){
    int r = q / CH, s = q - r*CH;
    u32 loff = lds_off(A1 + r*K1P) + s*16;
    const u16* g;
    if (s < 16)      g = hb  + (size_t)i*NODE_DIM + s*8;
    else if (s < 32) g = hb  + (size_t)sdst[r]*NODE_DIM + (s-16)*8;
    else             g = tail + (size_t)(i*DEG + r)*TAILW + (s-32)*8;
    async_copy16(loff, g);
  }
  async_wait0();
  __syncthreads();

  // GEMM1: (64 x 352) x (352 x 128)
  v8f acc[4];
#pragma unroll
  for (int mt = 0; mt < 4; ++mt) acc[mt] = v8zero();
  for (int kt = 0; kt < K1P/32; ++kt){
    v16bf b = load_b_frag(Wb1 + (size_t)(wid*16)*K1P + kt*32, K1P, lane);
#pragma unroll
    for (int mt = 0; mt < 4; ++mt){
      v16bf a = load_a_frag(A1 + mt*16*K1P + kt*32, K1P, lane);
      acc[mt] = wmma_bf16(a, b, acc[mt]);
    }
  }
  int half = lane >> 4, n = wid*16 + (lane & 15);
  float bias1 = e1b[n];
#pragma unroll
  for (int mt = 0; mt < 4; ++mt){
#pragma unroll
    for (int r8 = 0; r8 < 8; ++r8){
      int row = mt*16 + r8 + 8*half;
      M1[row*NODE_DIM + n] = f2bf(siluf(acc[mt][r8] + bias1));
    }
  }
  __syncthreads();

  // GEMM2: (64 x 128) x (128 x 128)
#pragma unroll
  for (int mt = 0; mt < 4; ++mt) acc[mt] = v8zero();
  for (int kt = 0; kt < NODE_DIM/32; ++kt){
    v16bf b = load_b_frag(Wb2 + (size_t)(wid*16)*NODE_DIM + kt*32, NODE_DIM, lane);
#pragma unroll
    for (int mt = 0; mt < 4; ++mt){
      v16bf a = load_a_frag(M1 + mt*16*NODE_DIM + kt*32, NODE_DIM, lane);
      acc[mt] = wmma_bf16(a, b, acc[mt]);
    }
  }
  float bias2 = e2b[n];
#pragma unroll
  for (int mt = 0; mt < 4; ++mt){
#pragma unroll
    for (int r8 = 0; r8 < 8; ++r8){
      int row = mt*16 + r8 + 8*half;
      M2[row*NODE_DIM + n] = siluf(acc[mt][r8] + bias2);
    }
  }
  __syncthreads();

  if (tid < 64){
    float s = 0.0f;
    if (tid < DEG){
      float a0 = attb[0];
      for (int c = 0; c < NODE_DIM; ++c) a0 += attW[c]*M2[tid*NODE_DIM + c];
      s = 1.0f / (1.0f + expf(-a0));
    }
    s_scale[tid] = s;
  }
  __syncthreads();

  if (tid < NODE_DIM){
    float sum = 0.0f;
    for (int r = 0; r < DEG; ++r) sum += M2[r*NODE_DIM + tid]*s_scale[r];
    agg[(size_t)i*NODE_DIM + tid] = sum;
  }
}

// ---------------- node update ----------------
__global__ void __launch_bounds__(128) node_update_kernel(float* __restrict__ h, u16* __restrict__ hb,
    const float* __restrict__ agg, const float* __restrict__ W1, const float* __restrict__ b1,
    const float* __restrict__ W2, const float* __restrict__ b2){
  __shared__ float sh[NODE_DIM], sa[NODE_DIM], so[NODE_DIM];
  int i = blockIdx.x, t = threadIdx.x;
  sh[t] = h[(size_t)i*NODE_DIM + t];
  sa[t] = agg[(size_t)i*NODE_DIM + t];
  __syncthreads();
  float o = b1[t];
  for (int k = 0; k < NODE_DIM; ++k) o += W1[(size_t)t*2*NODE_DIM + k]*sh[k];
  for (int k = 0; k < NODE_DIM; ++k) o += W1[(size_t)t*2*NODE_DIM + NODE_DIM + k]*sa[k];
  so[t] = siluf(o);
  __syncthreads();
  float v = sh[t] + b2[t];
  for (int k = 0; k < NODE_DIM; ++k) v += W2[(size_t)t*NODE_DIM + k]*so[k];
  h[(size_t)i*NODE_DIM + t]  = v;
  hb[(size_t)i*NODE_DIM + t] = f2bf(v);
}

// ---------------- energy head ----------------
__global__ void __launch_bounds__(256) energy_kernel(const u16* __restrict__ hb,
    const float* __restrict__ dm, const u16* __restrict__ W1b,
    const float* __restrict__ lng, const float* __restrict__ lnb, const float* __restrict__ W2,
    float* __restrict__ scal){
  __shared__ __align__(16) char smem[64*KEP*2];   // 36864; f32 OUT overlays after GEMM
  __shared__ float rsum[64], rcnt[64];
  u16*   A   = (u16*)smem;
  float* OUT = (float*)smem;
  int tid = threadIdx.x, wid = tid >> 5, lane = tid & 31;
  long p0 = (long)blockIdx.x*64;

  __builtin_prefetch(W1b + (size_t)(wid*16)*KEP, 0, 0);

  // async-stage h_rec | h_lig (32 x 16B chunks per row); zero invalid rows
  for (int q = tid; q < 64*32; q += 256){
    int r = q >> 5, s = q & 31;
    long p = p0 + r;
    u16* rowp = A + r*KEP;
    if (p < NPAIR){
      int ri = (int)(p / N_LIG), li = (int)(p - (long)ri*N_LIG);
      const u16* g = (s < 16) ? (hb + (size_t)ri*NODE_DIM + s*8)
                              : (hb + (size_t)(N_REC+li)*NODE_DIM + (s-16)*8);
      async_copy16(lds_off(rowp) + s*16, g);
    } else {
      u32* z = (u32*)rowp + s*4;
      z[0]=0; z[1]=0; z[2]=0; z[3]=0;
    }
  }
  // tail: cols 256..271 = [D_bf16, zeros]
  if (tid < 64){
    long p = p0 + tid;
    u32* tp = (u32*)(A + tid*KEP + 2*NODE_DIM);
    u32 d0 = 0;
    if (p < NPAIR){
      int ri = (int)(p / N_LIG), li = (int)(p - (long)ri*N_LIG);
      d0 = (u32)f2bf(dm[(size_t)ri*NTOT + N_REC + li]);
    }
    tp[0] = d0;
#pragma unroll
    for (int z = 1; z < 8; ++z) tp[z] = 0;
  }
  async_wait0();
  __syncthreads();

  v8f acc[4];
#pragma unroll
  for (int mt = 0; mt < 4; ++mt) acc[mt] = v8zero();
  for (int kt = 0; kt < KEP/32; ++kt){
    v16bf b = load_b_frag(W1b + (size_t)(wid*16)*KEP + kt*32, KEP, lane);
#pragma unroll
    for (int mt = 0; mt < 4; ++mt){
      v16bf a = load_a_frag(A + mt*16*KEP + kt*32, KEP, lane);
      acc[mt] = wmma_bf16(a, b, acc[mt]);
    }
  }
  __syncthreads();   // all reads of A done -> overlay with OUT

  int half = lane >> 4, n = wid*16 + (lane & 15);
#pragma unroll
  for (int mt = 0; mt < 4; ++mt){
#pragma unroll
    for (int r8 = 0; r8 < 8; ++r8){
      OUT[(mt*16 + r8 + 8*half)*NODE_DIM + n] = acc[mt][r8];
    }
  }
  __syncthreads();

  if (tid < 64){
    long p = p0 + tid;
    float s = 0.0f, cm = 0.0f;
    if (p < NPAIR){
      int ri = (int)(p / N_LIG), li = (int)(p - (long)ri*N_LIG);
      const float* rowv = OUT + tid*NODE_DIM;
      float mu = 0.0f;
      for (int c = 0; c < NODE_DIM; ++c) mu += rowv[c];
      mu *= (1.0f/NODE_DIM);
      float var = 0.0f;
      for (int c = 0; c < NODE_DIM; ++c){ float d = rowv[c]-mu; var += d*d; }
      var *= (1.0f/NODE_DIM);
      float inv = rsqrtf(var + 1e-5f);
      float e = 0.0f;
      for (int c = 0; c < NODE_DIM; ++c){
        float xn = (rowv[c]-mu)*inv*lng[c] + lnb[c];
        e += siluf(xn)*W2[c];
      }
      float D = dm[(size_t)ri*NTOT + N_REC + li];
      if (D < 100.0f){ s = e; cm = 1.0f; }
    }
    rsum[tid] = s; rcnt[tid] = cm;
  }
  __syncthreads();
  if (tid == 0){
    float s = 0.0f, c = 0.0f;
    for (int k = 0; k < 64; ++k){ s += rsum[k]; c += rcnt[k]; }
    atomicAdd(&scal[0], s);
    atomicAdd(&scal[1], c);
  }
}

__global__ void finalize_kernel(const float* __restrict__ scal, float* __restrict__ out){
  if (blockIdx.x == 0 && threadIdx.x == 0) out[0] = scal[0] / fmaxf(scal[1], 1.0f);
}

// ---------------- host side ----------------
struct ParamPtrs {
  const float *energy_W1, *energy_W2, *ln_b, *ln_g;
  const float *attW[4], *attb[4], *e1W[4], *e1b[4], *e2W[4], *e2b[4];
  const float *n1W[4], *n1b[4], *n2W[4], *n2b[4];
  const float *posW, *seW, *spW;
};

static void resolve_params(void* const* d_in, int n_in, ParamPtrs& P){
  if (n_in >= 52){
    int k = 5;
    P.energy_W1 = (const float*)d_in[k++];
    P.energy_W2 = (const float*)d_in[k++];
    P.ln_b      = (const float*)d_in[k++];
    P.ln_g      = (const float*)d_in[k++];
    for (int L = 0; L < 4; ++L){
      P.attW[L] = (const float*)d_in[k++]; P.attb[L] = (const float*)d_in[k++];
      P.e1W[L]  = (const float*)d_in[k++]; P.e1b[L]  = (const float*)d_in[k++];
      P.e2W[L]  = (const float*)d_in[k++]; P.e2b[L]  = (const float*)d_in[k++];
      P.n1W[L]  = (const float*)d_in[k++]; P.n1b[L]  = (const float*)d_in[k++];
      P.n2W[L]  = (const float*)d_in[k++]; P.n2b[L]  = (const float*)d_in[k++];
    }
    P.posW = (const float*)d_in[k++];
    P.seW  = (const float*)d_in[k++];
    P.spW  = (const float*)d_in[k++];
  } else {
    const float* base = (const float*)d_in[n_in - 1];
    size_t o = 0;
    auto take = [&](size_t cnt)->const float*{ const float* p = base + o; o += cnt; return p; };
    P.energy_W1 = take(128*257); P.energy_W2 = take(128); P.ln_b = take(128); P.ln_g = take(128);
    for (int L = 0; L < 4; ++L){
      P.attW[L] = take(128);     P.attb[L] = take(1);
      P.e1W[L]  = take(128*321); P.e1b[L]  = take(128);
      P.e2W[L]  = take(128*128); P.e2b[L]  = take(128);
      P.n1W[L]  = take(128*256); P.n1b[L]  = take(128);
      P.n2W[L]  = take(128*128); P.n2b[L]  = take(128);
    }
    P.posW = take(64*66); P.seW = take(128*1280); P.spW = take(64*100);
  }
}

extern "C" void kernel_launch(void* const* d_in, const int* in_sizes, int n_in,
                              void* d_out, int out_size, void* d_ws, size_t ws_size,
                              hipStream_t stream){
  (void)in_sizes; (void)out_size; (void)ws_size;
  const float* rec_x   = (const float*)d_in[0];
  const float* lig_x   = (const float*)d_in[1];
  const float* rec_pos = (const float*)d_in[2];
  const float* lig_pos = (const float*)d_in[3];
  const float* pm      = (const float*)d_in[4];
  ParamPtrs P; resolve_params(d_in, n_in, P);

  char* ws = (char*)d_ws;
  size_t off = 0;
  auto alloc = [&](size_t bytes)->char*{
    char* p = ws + off; off = (off + bytes + 255) & ~(size_t)255; return p;
  };
  u16*   xb    = (u16*)  alloc((size_t)NTOT*LM_DIM*2);
  u16*   WseB  = (u16*)  alloc((size_t)NODE_DIM*LM_DIM*2);
  u16*   Wb1   = (u16*)  alloc((size_t)4*NODE_DIM*K1P*2);
  u16*   Wb2   = (u16*)  alloc((size_t)4*NODE_DIM*NODE_DIM*2);
  u16*   EW1b  = (u16*)  alloc((size_t)NODE_DIM*KEP*2);
  float* ntb   = (float*)alloc((size_t)NTOT*3*4);
  float* cab   = (float*)alloc((size_t)NTOT*3*4);
  float* cbb   = (float*)alloc((size_t)NTOT*3*4);
  float* dm    = (float*)alloc((size_t)NTOT*NTOT*4);
  int*   dsti  = (int*)  alloc((size_t)NEDGE*4);
  float* ea    = (float*)alloc((size_t)NEDGE*EDGE_DIM*4);
  float* radl  = (float*)alloc((size_t)NEDGE*4);
  u16*   tailb = (u16*)  alloc((size_t)NEDGE*TAILW*2);
  float* h     = (float*)alloc((size_t)NTOT*NODE_DIM*4);
  u16*   hb    = (u16*)  alloc((size_t)NTOT*NODE_DIM*2);
  float* agg   = (float*)alloc((size_t)NTOT*NODE_DIM*4);
  float* scal  = (float*)alloc(256);

  init_scalars_kernel<<<1, 32, 0, stream>>>(scal);

  auto conv = [&](const float* src, u16* dst, int R, int C, int Cpad){
    int nb = (R*Cpad + 255)/256;
    convert_pad_kernel<<<nb, 256, 0, stream>>>(src, dst, R, C, Cpad);
  };
  conv(rec_x, xb,                        N_REC, LM_DIM, LM_DIM);
  conv(lig_x, xb + (size_t)N_REC*LM_DIM, N_LIG, LM_DIM, LM_DIM);
  conv(P.seW, WseB, NODE_DIM, LM_DIM, LM_DIM);
  for (int L = 0; L < 4; ++L){
    conv(P.e1W[L], Wb1 + (size_t)L*NODE_DIM*K1P,      NODE_DIM, K1,       K1P);
    conv(P.e2W[L], Wb2 + (size_t)L*NODE_DIM*NODE_DIM, NODE_DIM, NODE_DIM, NODE_DIM);
  }
  conv(P.energy_W1, EW1b, NODE_DIM, KE, KEP);

  prep_geom_kernel<<<(NTOT + 255)/256, 256, 0, stream>>>(rec_pos, lig_pos, ntb, cab, cbb);
  dm_kernel<<<(NTOT*NTOT + 255)/256, 256, 0, stream>>>(cab, dm);
  knn_sample_kernel<<<NTOT, 256, 0, stream>>>(dm, dsti);
  edge_attr_kernel<<<NEDGE, 64, 0, stream>>>(dsti, ntb, cab, cbb, pm, P.spW, P.posW, ea, radl);
  pack_tail_kernel<<<NEDGE, 96, 0, stream>>>(radl, ea, tailb);

  embed_kernel<<<(NTOT + 63)/64, 256, 0, stream>>>(xb, WseB, h, hb);

  for (int L = 0; L < 4; ++L){
    egcl_edge_kernel<<<NTOT, 256, 0, stream>>>(hb, dsti, tailb,
        Wb1 + (size_t)L*NODE_DIM*K1P, P.e1b[L],
        Wb2 + (size_t)L*NODE_DIM*NODE_DIM, P.e2b[L],
        P.attW[L], P.attb[L], agg);
    node_update_kernel<<<NTOT, 128, 0, stream>>>(h, hb, agg,
        P.n1W[L], P.n1b[L], P.n2W[L], P.n2b[L]);
  }

  energy_kernel<<<(NPAIR + 63)/64, 256, 0, stream>>>(hb, dm, EW1b,
      P.ln_g, P.ln_b, P.energy_W2, scal);
  finalize_kernel<<<1, 32, 0, stream>>>(scal, (float*)d_out);
}
